// TransformerEncoderLayer_CustomAttn_57690000720219
// MI455X (gfx1250) — matrix operand order repaired and checked by reading, not compiled, not run on hardware
//
#include <hip/hip_runtime.h>
#include <stdint.h>

// ---------------------------------------------------------------------------
// Problem constants (from reference)
// ---------------------------------------------------------------------------
#define S_DIM 2048
#define B_DIM 8
#define E_DIM 512
#define FF_DIM 2048

// ---------------------------------------------------------------------------
// bf16 helpers (raw ushort storage, round-to-nearest-even)
// ---------------------------------------------------------------------------
__device__ __forceinline__ unsigned short f2bf(float f) {
    union { float f; unsigned u; } c; c.f = f;
    unsigned u = c.u;
    return (unsigned short)((u + 0x7FFFu + ((u >> 16) & 1u)) >> 16);
}

typedef __attribute__((ext_vector_type(16))) __bf16 bf16x16;
typedef __attribute__((ext_vector_type(8)))  float  f32x8;
typedef __attribute__((ext_vector_type(4)))  unsigned int u32x4;
typedef __attribute__((ext_vector_type(4)))  int i32x4;

union Frag { bf16x16 v; unsigned u[8]; };

// ---------------------------------------------------------------------------
// CDNA5 async global->LDS copy (ASYNCcnt-tracked), with sync b128 fallback.
// Builtin signature (from probe diagnostic): (global int4*, local int4*, Ii, Ii)
// ---------------------------------------------------------------------------
#if defined(__AMDGCN__) && __has_builtin(__builtin_amdgcn_global_load_async_to_lds_b128) && \
    __has_builtin(__builtin_amdgcn_s_wait_asynccnt)
#define USE_ASYNC_LDS 1
typedef __attribute__((address_space(1))) i32x4 gas_i32x4;
typedef __attribute__((address_space(3))) i32x4 las_i32x4;
#else
#define USE_ASYNC_LDS 0
#endif

__device__ __forceinline__ void copy16_g2l(const unsigned short* g, unsigned short* l) {
#if USE_ASYNC_LDS
    __builtin_amdgcn_global_load_async_to_lds_b128(
        (gas_i32x4*)(uintptr_t)g,
        (las_i32x4*)(uintptr_t)l,
        0, 0);
#else
    *(u32x4*)l = *(const u32x4*)g;
#endif
}

__device__ __forceinline__ void stage_join() {
#if USE_ASYNC_LDS
    __builtin_amdgcn_s_wait_asynccnt(0);
#endif
    __syncthreads();
}

// ---------------------------------------------------------------------------
// elementwise fp32 -> bf16 cast
// ---------------------------------------------------------------------------
__global__ __launch_bounds__(256) void cast_bf16_kernel(const float* __restrict__ in,
                                                        unsigned short* __restrict__ out,
                                                        long n) {
    long i = (long)blockIdx.x * blockDim.x + threadIdx.x;
    long stride = (long)gridDim.x * blockDim.x;
    for (; i < n; i += stride) out[i] = f2bf(in[i]);
}

// ---------------------------------------------------------------------------
// L1-normalize pos_embed over channel dim + transpose:
//   peT[b][s][e] = pos[b][e][s] / max(sum_e |pos[b][e][s]|, 1e-12)
// One wave per (b,s) column; 8 waves per block.
// ---------------------------------------------------------------------------
__global__ __launch_bounds__(256) void l1norm_transpose_kernel(const float* __restrict__ pos,
                                                               unsigned short* __restrict__ peT) {
    const int wid  = threadIdx.x >> 5;
    const int lane = threadIdx.x & 31;
    const int col  = blockIdx.x * 8 + wid;      // 0 .. B*S-1
    const int b = col >> 11;                    // /S
    const int s = col & (S_DIM - 1);
    const float* base = pos + (size_t)b * E_DIM * S_DIM + s;
    float vals[16];
    float sum = 0.0f;
#pragma unroll
    for (int i = 0; i < 16; ++i) {
        int e = i * 32 + lane;
        vals[i] = base[(size_t)e * S_DIM];
        sum += fabsf(vals[i]);
    }
#pragma unroll
    for (int off = 16; off; off >>= 1) sum += __shfl_xor(sum, off, 32);
    const float r = 1.0f / fmaxf(sum, 1e-12f);
    unsigned short* out = peT + ((size_t)b * S_DIM + s) * E_DIM;
#pragma unroll
    for (int i = 0; i < 16; ++i) out[i * 32 + lane] = f2bf(vals[i] * r);
}

// ---------------------------------------------------------------------------
// Row softmax over fp32 scores (length S), writing bf16 IN PLACE into the
// leading half of each fp32 row (attn rows then live at stride 2*S ushorts).
// ---------------------------------------------------------------------------
__global__ __launch_bounds__(256) void softmax_bf16_inplace_kernel(float* __restrict__ scores) {
    __shared__ float sred[8];
    const int tid  = threadIdx.x;
    const int wid  = tid >> 5;
    const int lane = tid & 31;
    float* base = scores + (size_t)blockIdx.x * S_DIM;

    float vals[8];
    float m = -3.4e38f;
#pragma unroll
    for (int i = 0; i < 8; ++i) { vals[i] = base[tid + i * 256]; m = fmaxf(m, vals[i]); }
#pragma unroll
    for (int off = 16; off; off >>= 1) m = fmaxf(m, __shfl_xor(m, off, 32));
    if (lane == 0) sred[wid] = m;
    __syncthreads();
    if (tid == 0) {
        float t = sred[0];
#pragma unroll
        for (int i = 1; i < 8; ++i) t = fmaxf(t, sred[i]);
        sred[0] = t;
    }
    __syncthreads();
    m = sred[0];
    __syncthreads();

    float sum = 0.0f;
#pragma unroll
    for (int i = 0; i < 8; ++i) { vals[i] = expf(vals[i] - m); sum += vals[i]; }
#pragma unroll
    for (int off = 16; off; off >>= 1) sum += __shfl_xor(sum, off, 32);
    if (lane == 0) sred[wid] = sum;
    __syncthreads();
    if (tid == 0) {
        float t = 0.0f;
#pragma unroll
        for (int i = 0; i < 8; ++i) t += sred[i];
        sred[0] = t;
    }
    __syncthreads();
    const float r = 1.0f / sred[0];

    unsigned short* o = (unsigned short*)base;   // in-place bf16 (row stride 2*S)
#pragma unroll
    for (int i = 0; i < 8; ++i) o[tid + i * 256] = f2bf(vals[i] * r);
}

// ---------------------------------------------------------------------------
// Fused residual + LayerNorm over rows of E=512.
// ---------------------------------------------------------------------------
__global__ __launch_bounds__(256) void resid_ln_kernel(const float* __restrict__ a,
                                                       const float* __restrict__ b,
                                                       const float* __restrict__ g,
                                                       const float* __restrict__ beta,
                                                       float* __restrict__ outF,
                                                       unsigned short* __restrict__ outH) {
    __shared__ float s1[8], s2[8];
    const int tid  = threadIdx.x;
    const int wid  = tid >> 5;
    const int lane = tid & 31;
    const size_t base = (size_t)blockIdx.x * E_DIM;

    float x0 = a[base + tid]       + b[base + tid];
    float x1 = a[base + tid + 256] + b[base + tid + 256];
    float sum = x0 + x1;
    float sq  = x0 * x0 + x1 * x1;
#pragma unroll
    for (int off = 16; off; off >>= 1) {
        sum += __shfl_xor(sum, off, 32);
        sq  += __shfl_xor(sq,  off, 32);
    }
    if (lane == 0) { s1[wid] = sum; s2[wid] = sq; }
    __syncthreads();
    if (tid == 0) {
        float t1 = 0.0f, t2 = 0.0f;
#pragma unroll
        for (int i = 0; i < 8; ++i) { t1 += s1[i]; t2 += s2[i]; }
        s1[0] = t1; s2[0] = t2;
    }
    __syncthreads();
    const float mean = s1[0] * (1.0f / E_DIM);
    const float var  = s2[0] * (1.0f / E_DIM) - mean * mean;
    const float rstd = rsqrtf(var + 1e-5f);

    float y0 = (x0 - mean) * rstd * g[tid]       + beta[tid];
    float y1 = (x1 - mean) * rstd * g[tid + 256] + beta[tid + 256];
    outF[base + tid]       = y0;
    outF[base + tid + 256] = y1;
    if (outH) {
        outH[base + tid]       = f2bf(y0);
        outH[base + tid + 256] = f2bf(y1);
    }
}

// ---------------------------------------------------------------------------
// Tiled bf16 WMMA GEMM:  C[M,N] = A[M,K] (row, lda) * B[N,K]^T (row, ldb)
// fp32 accumulate via v_wmma_f32_16x16x32_bf16.
// Block: 256 threads = 8 waves (2 m x 4 n); block tile 64x256, wave tile
// 32x64 (2x4 WMMA frags -> 8 WMMA per K-step of 32). LDS double-buffered,
// staged with async global->LDS b128 copies when available.
// Epilogue MODE:
//   0 : out fp32 row-major (+batch)            [scores]
//   1 : out bf16 row-major (+batch)            [ctx]
//   2 : scatter bf16 to vT[b][n][s], +bias     [V projection]
//   3 : scatter fp32 to [s][b][n], +bias       [O projection]
//   4 : bias + relu -> bf16 row-major          [FFN1]
//   5 : bias -> fp32 row-major                 [FFN2]
// ---------------------------------------------------------------------------
#define TILE_M 64
#define TILE_N 256
#define TILE_K 32
#define LDSS   40   // padded row stride (ushorts); 80 B = 5*16 keeps b128 align

template <int MODE>
__global__ __launch_bounds__(256) void gemm_bt_kernel(
    const unsigned short* __restrict__ Abase, int lda, long strideA,
    const unsigned short* __restrict__ Bbase, int ldb, long strideB,
    float* __restrict__ outF, unsigned short* __restrict__ outH,
    int ldc, long strideC,
    const float* __restrict__ bias,
    int M, int N, int K, int Sdim, int Bdim) {

    __shared__ unsigned short As[2][TILE_M * LDSS];
    __shared__ unsigned short Bs[2][TILE_N * LDSS];

    const int tid    = threadIdx.x;
    const int wid    = tid >> 5;
    const int lane   = tid & 31;
    const int wave_m = wid & 1;
    const int wave_n = wid >> 1;
    const int bz     = blockIdx.z;
    const int m0     = blockIdx.y * TILE_M;
    const int n0     = blockIdx.x * TILE_N;
    const int half   = lane >> 4;
    const int lm     = lane & 15;

    const unsigned short* A  = Abase + (size_t)bz * strideA;
    const unsigned short* Bp = Bbase + (size_t)bz * strideB;

    f32x8 acc[2][4] = {};

    // stage one K-tile (16-byte chunks; A: 256 chunks, B: 1024 chunks)
    auto stage = [&](int k0, int buf) {
        {
            const int r  = tid >> 2;            // 0..63
            const int ko = (tid & 3) << 3;      // 0,8,16,24
            copy16_g2l(A + (size_t)(m0 + r) * lda + k0 + ko,
                       &As[buf][r * LDSS + ko]);
        }
#pragma unroll
        for (int i = 0; i < 4; ++i) {
            const int c  = tid + i * 256;
            const int r  = c >> 2;              // 0..255
            const int ko = (c & 3) << 3;
            copy16_g2l(Bp + (size_t)(n0 + r) * ldb + k0 + ko,
                       &Bs[buf][r * LDSS + ko]);
        }
    };

    stage(0, 0);
    int buf = 0;
    for (int k0 = 0; k0 < K; k0 += TILE_K) {
        stage_join();                            // staging of `buf` visible to all
        if (k0 + TILE_K < K) stage(k0 + TILE_K, buf ^ 1);   // overlap next tile

        // build fragments (ISA 7.12.2 layouts) and issue 8 WMMAs
        Frag a[2], b[4];
#pragma unroll
        for (int i = 0; i < 2; ++i) {
            const unsigned short* ap = &As[buf][(wave_m * 32 + i * 16 + lm) * LDSS];
#pragma unroll
            for (int v = 0; v < 4; ++v)
                a[i].u[v] = *(const unsigned*)(ap + half * 8 + 2 * v);
#pragma unroll
            for (int v = 4; v < 8; ++v)
                a[i].u[v] = *(const unsigned*)(ap + 16 + half * 8 + 2 * (v - 4));
        }
#pragma unroll
        for (int j = 0; j < 4; ++j) {
            const unsigned short* bp = &Bs[buf][(wave_n * 64 + j * 16 + lm) * LDSS];
#pragma unroll
            for (int v = 0; v < 4; ++v)
                b[j].u[v] = *(const unsigned*)(bp + half * 8 + 2 * v);
#pragma unroll
            for (int v = 4; v < 8; ++v)
                b[j].u[v] = *(const unsigned*)(bp + 16 + half * 8 + 2 * (v - 4));
        }
#pragma unroll
        for (int i = 0; i < 2; ++i)
#pragma unroll
            for (int j = 0; j < 4; ++j)
                acc[i][j] = __builtin_amdgcn_wmma_f32_16x16x32_bf16(
                    false, a[i].v, false, b[j].v, (short)0, acc[i][j], false, false);
        buf ^= 1;
    }

    // epilogue (C/D layout: lane lm = N, VGPR r -> M = r + 8*half)
#pragma unroll
    for (int i = 0; i < 2; ++i) {
#pragma unroll
        for (int j = 0; j < 4; ++j) {
            const int gm_base = m0 + wave_m * 32 + i * 16 + half * 8;
            const int gn      = n0 + wave_n * 64 + j * 16 + lm;
            float bv_ = 0.0f;
            if (MODE >= 2) bv_ = bias[gn];
#pragma unroll
            for (int r = 0; r < 8; ++r) {
                const int gm = gm_base + r;
                float x = acc[i][j][r] + bv_;
                if (MODE == 4) x = fmaxf(x, 0.0f);
                if (MODE == 0) {
                    (outF + (size_t)bz * strideC)[(size_t)gm * ldc + gn] = x;
                } else if (MODE == 1) {
                    (outH + (size_t)bz * strideC)[(size_t)gm * ldc + gn] = f2bf(x);
                } else if (MODE == 2) {                 // rows are (s*B + b); vT[b][n][s]
                    int s  = gm / Bdim;
                    int bb = gm - s * Bdim;
                    outH[((size_t)bb * N + gn) * Sdim + s] = f2bf(x);
                } else if (MODE == 3) {                 // rows are (b*S + s); out[s][b][n]
                    int bb = gm / Sdim;
                    int s  = gm - bb * Sdim;
                    outF[((size_t)s * Bdim + bb) * N + gn] = x;
                } else if (MODE == 4) {
                    outH[(size_t)gm * ldc + gn] = f2bf(x);
                } else {                                // 5
                    outF[(size_t)gm * ldc + gn] = x;
                }
            }
        }
    }
}

// ---------------------------------------------------------------------------
// Host launcher
// ---------------------------------------------------------------------------
extern "C" void kernel_launch(void* const* d_in, const int* in_sizes, int n_in,
                              void* d_out, int out_size, void* d_ws, size_t ws_size,
                              hipStream_t stream) {
    (void)in_sizes; (void)n_in; (void)out_size; (void)ws_size;
    const float* src = (const float*)d_in[0];
    const float* pos = (const float*)d_in[1];
    const float* Wv  = (const float*)d_in[2];
    const float* bv  = (const float*)d_in[3];
    const float* Wo  = (const float*)d_in[4];
    const float* bo  = (const float*)d_in[5];
    const float* W1  = (const float*)d_in[6];
    const float* b1  = (const float*)d_in[7];
    const float* W2  = (const float*)d_in[8];
    const float* b2  = (const float*)d_in[9];
    const float* g1  = (const float*)d_in[10];
    const float* be1 = (const float*)d_in[11];
    const float* g2  = (const float*)d_in[12];
    const float* be2 = (const float*)d_in[13];
    float* out = (float*)d_out;

    const long SBE = (long)S_DIM * B_DIM * E_DIM;
    const long SS  = (long)S_DIM * S_DIM;

    // workspace layout
    char* w = (char*)d_ws;
    size_t off = 0;
    unsigned short* ws_peT  = (unsigned short*)(w + off); off += 16777216UL;   // [B,S,E] bf16
    float*          ws_sc   = (float*)(w + off);          off += 134217728UL;  // [B,S,S] f32 (bf16 in-place)
    unsigned short* ws_vT   = (unsigned short*)(w + off); off += 16777216UL;   // [B,E,S] bf16
    unsigned short* ws_srcb = (unsigned short*)(w + off); off += 16777216UL;   // [S,B,E] bf16
    unsigned short* ws_ctx  = (unsigned short*)(w + off); off += 16777216UL;   // [B,S,E] bf16
    float*          ws_x    = (float*)(w + off);          off += 33554432UL;   // [S,B,E] f32
    unsigned short* ws_xb   = (unsigned short*)(w + off); off += 16777216UL;   // [S,B,E] bf16
    unsigned short* ws_h    = (unsigned short*)(w + off); off += 67108864UL;   // [S*B,FF] bf16
    float*          ws_tmp  = (float*)(w + off);          off += 33554432UL;   // [S,B,E] f32
    unsigned short* ws_Wv   = (unsigned short*)(w + off); off += 524288UL;
    unsigned short* ws_Wo   = (unsigned short*)(w + off); off += 524288UL;
    unsigned short* ws_W1   = (unsigned short*)(w + off); off += 2097152UL;
    unsigned short* ws_W2   = (unsigned short*)(w + off); off += 2097152UL;

    // 1) casts to bf16
    cast_bf16_kernel<<<4096, 256, 0, stream>>>(src, ws_srcb, SBE);
    cast_bf16_kernel<<<512,  256, 0, stream>>>(Wv, ws_Wv, (long)E_DIM * E_DIM);
    cast_bf16_kernel<<<512,  256, 0, stream>>>(Wo, ws_Wo, (long)E_DIM * E_DIM);
    cast_bf16_kernel<<<2048, 256, 0, stream>>>(W1, ws_W1, (long)FF_DIM * E_DIM);
    cast_bf16_kernel<<<2048, 256, 0, stream>>>(W2, ws_W2, (long)E_DIM * FF_DIM);

    // 2) L1 normalize + transpose pos_embed -> peT [B,S,E]
    l1norm_transpose_kernel<<<(B_DIM * S_DIM) / 8, 256, 0, stream>>>(pos, ws_peT);

    // 3) scores[b] = peT[b] @ peT[b]^T   (M=N=2048, K=512) -> fp32
    gemm_bt_kernel<0><<<dim3(S_DIM / TILE_N, S_DIM / TILE_M, B_DIM), 256, 0, stream>>>(
        ws_peT, E_DIM, (long)S_DIM * E_DIM,
        ws_peT, E_DIM, (long)S_DIM * E_DIM,
        ws_sc, nullptr, S_DIM, SS, nullptr,
        S_DIM, S_DIM, E_DIM, S_DIM, B_DIM);

    // 4) softmax rows -> bf16 in place (attn rows at stride 2*S ushorts)
    softmax_bf16_inplace_kernel<<<B_DIM * S_DIM, 256, 0, stream>>>(ws_sc);

    // 5) v = src @ Wv^T + bv, scattered to vT[b][e][s]   (M=S*B, N=E, K=E)
    gemm_bt_kernel<2><<<dim3(E_DIM / TILE_N, (S_DIM * B_DIM) / TILE_M, 1), 256, 0, stream>>>(
        ws_srcb, E_DIM, 0, ws_Wv, E_DIM, 0,
        nullptr, ws_vT, 0, 0, bv,
        S_DIM * B_DIM, E_DIM, E_DIM, S_DIM, B_DIM);

    // 6) ctx[b] = attn[b] @ vT[b]^T -> bf16 [B,S,E]   (M=S, N=E, K=S)
    gemm_bt_kernel<1><<<dim3(E_DIM / TILE_N, S_DIM / TILE_M, B_DIM), 256, 0, stream>>>(
        (const unsigned short*)ws_sc, 2 * S_DIM, SS * 2,
        ws_vT, S_DIM, (long)E_DIM * S_DIM,
        nullptr, ws_ctx, E_DIM, (long)S_DIM * E_DIM, nullptr,
        S_DIM, E_DIM, S_DIM, S_DIM, B_DIM);

    // 7) attn_out = ctx @ Wo^T + bo, scattered to [S,B,E] fp32
    gemm_bt_kernel<3><<<dim3(E_DIM / TILE_N, (S_DIM * B_DIM) / TILE_M, 1), 256, 0, stream>>>(
        ws_ctx, E_DIM, 0, ws_Wo, E_DIM, 0,
        ws_tmp, nullptr, 0, 0, bo,
        S_DIM * B_DIM, E_DIM, E_DIM, S_DIM, B_DIM);

    // 8) x = LN(src + attn_out) -> fp32 + bf16
    resid_ln_kernel<<<S_DIM * B_DIM, 256, 0, stream>>>(src, ws_tmp, g1, be1, ws_x, ws_xb);

    // 9) h = relu(x @ W1^T + b1) -> bf16   (M=S*B, N=FF, K=E)
    gemm_bt_kernel<4><<<dim3(FF_DIM / TILE_N, (S_DIM * B_DIM) / TILE_M, 1), 256, 0, stream>>>(
        ws_xb, E_DIM, 0, ws_W1, E_DIM, 0,
        nullptr, ws_h, FF_DIM, 0, b1,
        S_DIM * B_DIM, FF_DIM, E_DIM, S_DIM, B_DIM);

    // 10) ff = h @ W2^T + b2 -> fp32       (M=S*B, N=E, K=FF)
    gemm_bt_kernel<5><<<dim3(E_DIM / TILE_N, (S_DIM * B_DIM) / TILE_M, 1), 256, 0, stream>>>(
        ws_h, FF_DIM, 0, ws_W2, FF_DIM, 0,
        ws_tmp, nullptr, E_DIM, 0, b2,
        S_DIM * B_DIM, E_DIM, FF_DIM, S_DIM, B_DIM);

    // 11) out = LN(x + ff) -> d_out fp32 [S,B,E]
    resid_ln_kernel<<<S_DIM * B_DIM, 256, 0, stream>>>(ws_x, ws_tmp, g2, be2, out, nullptr);
}
